// Variance_Predictor_Block_4277787427514
// MI455X (gfx1250) — matrix operand order repaired and checked by reading, not compiled, not run on hardware
//
#include <hip/hip_runtime.h>
#include <hip/hip_bf16.h>

#define B_     32
#define E_     192
#define T_     1024
#define TP_    (T_ + 2)   // padded t dimension (zero row at 0 and T_+1)
#define C_     256
#define OUT_T_ 5120
#define BN_EPS 1e-5f

typedef __attribute__((ext_vector_type(16))) __bf16       v16bf;
typedef __attribute__((ext_vector_type(8)))  float        v8f;
typedef __attribute__((ext_vector_type(4)))  unsigned int u32x4;

union FragU { v16bf v; u32x4 q[2]; unsigned int u[8]; };

__device__ __forceinline__ v16bf load_frag(const __bf16* row, int off0, int off1) {
  FragU f;
  f.q[0] = *reinterpret_cast<const u32x4*>(row + off0);
  f.q[1] = *reinterpret_cast<const u32x4*>(row + off1);
  return f.v;
}
__device__ __forceinline__ float mishf(float x) {
  float sp = (x > 20.f) ? x : __logf(1.f + __expf(x));
  return x * tanhf(sp);
}

// ---------------------------------------------------------------------------
// Weight prepack: W [3][C][KIN][3] (p,c,e,k) f32  ->  Wp [3][3][C][KIN] bf16
// ---------------------------------------------------------------------------
template <int KIN>
__global__ void pack_w(const float* __restrict__ W, __bf16* __restrict__ Wp) {
  int i = blockIdx.x * blockDim.x + threadIdx.x;
  const int total = 3 * C_ * KIN * 3;
  if (i >= total) return;
  int k = i % 3;
  int e = (i / 3) % KIN;
  int c = (i / (3 * KIN)) % C_;
  int p = i / (3 * KIN * C_);
  Wp[(((size_t)p * 3 + k) * C_ + c) * KIN + e] = (__bf16)W[i];
}

// ---------------------------------------------------------------------------
// Zero the two pad rows (t = 0 and t = T+1) of a padded t-major buffer.
// ---------------------------------------------------------------------------
template <int KIN>
__global__ void zero_pad_rows(__bf16* __restrict__ Xp, int outer) {
  int i = blockIdx.x * blockDim.x + threadIdx.x;
  int total = outer * 2 * KIN;
  if (i >= total) return;
  int e = i % KIN;
  int r = (i / KIN) % 2;
  int o = i / (2 * KIN);
  Xp[((size_t)o * TP_ + (r ? (T_ + 1) : 0)) * KIN + e] = (__bf16)0.f;
}

// ---------------------------------------------------------------------------
// LDS-tiled transpose [outer][ROWS][T] f32 -> padded [outer][T+2][ROWS] bf16
// (data at rows 1..T), optional BN fold (p = outer_idx / bnPerOuter).
// Block (32,8), 32x32 tile, both global sides coalesced.
// ---------------------------------------------------------------------------
template <int ROWS>
__global__ __launch_bounds__(256) void transpose_bn_bf16(
    const float* __restrict__ X, __bf16* __restrict__ Y,
    const float* __restrict__ scale, const float* __restrict__ shift,
    int bnPerOuter) {
  __shared__ float tile[32][33];
  const int tx = threadIdx.x;  // 0..31
  const int ty = threadIdx.y;  // 0..7
  const int t0 = blockIdx.x * 32;
  const int r0 = blockIdx.y * 32;
  const int o  = blockIdx.z;
  const float* src = X + ((size_t)o * ROWS + r0) * T_ + t0;
#pragma unroll
  for (int j = 0; j < 4; ++j)
    tile[ty + j * 8][tx] = src[(size_t)(ty + j * 8) * T_ + tx];
  __syncthreads();
  const int r = r0 + tx;
  float sc = 1.f, sh = 0.f;
  if (scale) {
    int p = o / bnPerOuter;
    sc = scale[p * ROWS + r];
    sh = shift[p * ROWS + r];
  }
  __bf16* dst = Y + ((size_t)o * TP_ + t0 + 1) * ROWS + r0;
#pragma unroll
  for (int j = 0; j < 4; ++j) {
    float v = tile[tx][ty + j * 8] * sc + sh;
    dst[(size_t)(ty + j * 8) * ROWS + tx] = (__bf16)v;
  }
}

// ---------------------------------------------------------------------------
// Conv1d(K=3) as WMMA GEMM.
// X: padded t-major bf16 [outer][T+2][KIN] (outer = B or 3B), zero pad rows.
// W: [3][3][C][KIN] bf16.  Y: [3][B][C][T] f32 (mish applied) + BN stats.
// Each wave: 16(C) x 64(T) tile, 4 accumulators, A-fragment reused 4x.
// Block = 8 waves = 8 consecutive c-tiles sharing one t-group.
// Grid: (T/64, C/128, 3*B)
// ---------------------------------------------------------------------------
template <int KIN, int XPERP>
__global__ __launch_bounds__(256) void conv_gemm_wmma(
    const __bf16* __restrict__ X, const __bf16* __restrict__ W,
    const float* __restrict__ bias, float* __restrict__ Y,
    float* __restrict__ sum, float* __restrict__ sumsq) {
  const int lane  = threadIdx.x & 31;
  const int wave  = threadIdx.x >> 5;
  const int t0    = blockIdx.x * 64;
  const int c0    = blockIdx.y * 128 + wave * 16;
  const int p     = blockIdx.z / B_;
  const int b     = blockIdx.z % B_;
  const int n     = lane & 15;
  const int khalf = lane >> 4;

  v8f acc[4] = {};
  const __bf16* xbase =
      X + (size_t)(XPERP ? (p * B_ + b) : b) * TP_ * KIN;
  const int ka = khalf * 8;   // A: elems 0-7 at K=ka, elems 8-15 at K=16+ka
  const int kb = khalf * 8;   // B: elems 0-7 at K=kb, elems 8-15 at K=16+kb

#pragma unroll
  for (int k = 0; k < 3; ++k) {
    const __bf16* arow = W + (((size_t)(p * 3 + k) * C_ + c0 + n) * KIN);
    // padded row index: t + 1 = t0 + nt*16 + n + (k-1) + 1
    const __bf16* brow0 = xbase + (size_t)(t0 + n + k) * KIN;
#pragma unroll
    for (int ch = 0; ch < KIN / 32; ++ch) {
      v16bf afrag = load_frag(arow, ch * 32 + ka, ch * 32 + 16 + ka);
#pragma unroll
      for (int nt = 0; nt < 4; ++nt) {
        v16bf bfrag = load_frag(brow0 + (size_t)nt * 16 * KIN,
                                ch * 32 + kb, ch * 32 + 16 + kb);
        acc[nt] = __builtin_amdgcn_wmma_f32_16x16x32_bf16(
            false, afrag, false, bfrag, (short)0, acc[nt], false, false);
      }
    }
  }

  const int mb = khalf * 8;  // D: VGPR j -> row mb+j, col n (per n-tile)
#pragma unroll
  for (int j = 0; j < 8; ++j) {
    const int c = c0 + mb + j;
    const float bi = bias[p * C_ + c];
    float* yrow = Y + (((size_t)p * B_ + b) * C_ + c) * T_ + t0 + n;
    float s = 0.f, s2 = 0.f;
#pragma unroll
    for (int nt = 0; nt < 4; ++nt) {
      float v = mishf(acc[nt][j] + bi);
      yrow[nt * 16] = v;
      s  += v;
      s2 += v * v;
    }
#pragma unroll
    for (int off = 1; off < 16; off <<= 1) {
      s  += __shfl_xor(s,  off, 32);
      s2 += __shfl_xor(s2, off, 32);
    }
    if (n == 0) {
      atomicAdd(&sum[p * C_ + c],   s);
      atomicAdd(&sumsq[p * C_ + c], s2);
    }
  }
}

// ---------------------------------------------------------------------------
// BN finalize: per (p,c) scale/shift from accumulated sum/sumsq.
// ---------------------------------------------------------------------------
__global__ void bn_finalize(const float* __restrict__ sum,
                            const float* __restrict__ sq,
                            const float* __restrict__ g,
                            const float* __restrict__ be,
                            float* __restrict__ scale, float* __restrict__ shift) {
  int i = blockIdx.x * blockDim.x + threadIdx.x;
  if (i >= 3 * C_) return;
  const float invn = 1.f / (float)(B_ * T_);
  float m = sum[i] * invn;
  float v = sq[i] * invn - m * m;
  float sc = g[i] * rsqrtf(v + BN_EPS);
  scale[i] = sc;
  shift[i] = be[i] - m * sc;
}

// ---------------------------------------------------------------------------
// Pointwise conv: pred[p][b][t] = sum_c Wp[p][c]*(bn2(h1)) + bp[p]
// ---------------------------------------------------------------------------
__global__ __launch_bounds__(256) void final_pw(
    const float* __restrict__ H, const float* __restrict__ scale,
    const float* __restrict__ shift, const float* __restrict__ Wp,
    const float* __restrict__ bp, float* __restrict__ out_dur,
    float* __restrict__ out_f0, float* __restrict__ out_en) {
  int t = blockIdx.x * blockDim.x + threadIdx.x;
  int b = blockIdx.y;
  int p = blockIdx.z;
  const float* h = H + (((size_t)p * B_ + b) * C_) * T_ + t;
  float acc = bp[p];
#pragma unroll 4
  for (int c = 0; c < C_; ++c)
    acc += Wp[p * C_ + c] * (h[(size_t)c * T_] * scale[p * C_ + c] + shift[p * C_ + c]);
  float* o = (p == 0) ? out_dur : (p == 1 ? out_f0 : out_en);
  o[b * T_ + t] = acc;
}

// ---------------------------------------------------------------------------
// means staging: enc + masked f0/energy 9-tap embedding convs
// ---------------------------------------------------------------------------
__global__ __launch_bounds__(256) void emb_means(
    const float* __restrict__ enc, const float* __restrict__ mask,
    const float* __restrict__ f0, const float* __restrict__ en,
    const float* __restrict__ f0W, const float* __restrict__ f0b,
    const float* __restrict__ enW, const float* __restrict__ enb,
    float* __restrict__ outm) {
  int t = blockIdx.x * blockDim.x + threadIdx.x;
  int e = blockIdx.y;
  int b = blockIdx.z;
  float fa = f0b[e], ea = enb[e];
#pragma unroll
  for (int kk = 0; kk < 9; ++kk) {
    int tt = t + kk - 4;
    if (tt >= 0 && tt < T_) {
      fa += f0W[e * 9 + kk] * f0[b * T_ + tt];
      ea += enW[e * 9 + kk] * en[b * T_ + tt];
    }
  }
  size_t i = ((size_t)b * E_ + e) * T_ + t;
  outm[i] = enc[i] + (fa + ea) * mask[b * T_ + t];
}

// ---------------------------------------------------------------------------
// Length regulation: per-batch inclusive scan in LDS, scatter source index.
// ---------------------------------------------------------------------------
__global__ __launch_bounds__(1024) void lr_idx(const int* __restrict__ dur,
                                               int* __restrict__ idx) {
  __shared__ int cum[T_];
  int t = threadIdx.x;
  int b = blockIdx.x;
  cum[t] = dur[b * T_ + t];
  __syncthreads();
  for (int off = 1; off < T_; off <<= 1) {
    int add = (t >= off) ? cum[t - off] : 0;
    __syncthreads();
    cum[t] += add;
    __syncthreads();
  }
  int end = cum[t];
  int start = (t == 0) ? 0 : cum[t - 1];
  if (end > OUT_T_) end = OUT_T_;
  for (int o = start; o < end; ++o) idx[b * OUT_T_ + o] = t;
}

__global__ __launch_bounds__(256) void lr_gather(
    const float* __restrict__ meansS, const float* __restrict__ logstd,
    const int* __restrict__ idx, float* __restrict__ outM,
    float* __restrict__ outS) {
  size_t i = (size_t)blockIdx.x * blockDim.x + threadIdx.x;
  const size_t total = (size_t)B_ * E_ * OUT_T_;
  if (i >= total) return;
  int to = (int)(i % OUT_T_);
  int e  = (int)((i / OUT_T_) % E_);
  int b  = (int)(i / ((size_t)OUT_T_ * E_));
  int j = idx[b * OUT_T_ + to];
  size_t src = ((size_t)b * E_ + e) * T_ + j;
  outM[i] = meansS[src];
  outS[i] = logstd[src];
}

// ---------------------------------------------------------------------------
extern "C" void kernel_launch(void* const* d_in, const int* in_sizes, int n_in,
                              void* d_out, int out_size, void* d_ws, size_t ws_size,
                              hipStream_t stream) {
  const float* enc     = (const float*)d_in[0];
  const float* logstdp = (const float*)d_in[1];
  const float* mask    = (const float*)d_in[2];
  const float* logf0s  = (const float*)d_in[3];
  const float* energies= (const float*)d_in[4];
  const int*   dur     = (const int*)d_in[5];
  const float* P_W0    = (const float*)d_in[7];
  const float* P_b0    = (const float*)d_in[8];
  const float* P_g0    = (const float*)d_in[9];
  const float* P_be0   = (const float*)d_in[10];
  const float* P_W1    = (const float*)d_in[11];
  const float* P_b1    = (const float*)d_in[12];
  const float* P_g1    = (const float*)d_in[13];
  const float* P_be1   = (const float*)d_in[14];
  const float* P_Wp    = (const float*)d_in[15];
  const float* P_bp    = (const float*)d_in[16];
  const float* f0W     = (const float*)d_in[17];
  const float* f0b     = (const float*)d_in[18];
  const float* enW     = (const float*)d_in[19];
  const float* enb     = (const float*)d_in[20];

  // workspace layout (all 256B aligned)
  char* w = (char*)d_ws;
  const size_t OFF_XB0   = 0;                    // B*(T+2)*E bf16   = 12,607,488
  const size_t OFF_W0    = OFF_XB0 + 12607488;   // 3*3*C*E bf16     = 884,736
  const size_t OFF_W1    = OFF_W0 + 884736;      // 3*3*C*C bf16     = 1,179,648
  const size_t OFF_H     = OFF_W1 + 1179648;     // 3*B*C*T f32      = 100,663,296
  const size_t OFF_XB1   = OFF_H + 100663296;    // 3B*(T+2)*C bf16  = 50,429,952
  const size_t OFF_ST    = OFF_XB1 + 50429952;   // 8*768 f32        = 24,576
  const size_t OFF_MEANS = OFF_ST + 24576;       // B*E*T f32        = 25,165,824
  const size_t OFF_IDX   = OFF_MEANS + 25165824; // B*OUT_T i32      = 655,360

  __bf16* Xb0  = (__bf16*)(w + OFF_XB0);
  __bf16* Wp0  = (__bf16*)(w + OFF_W0);
  __bf16* Wp1  = (__bf16*)(w + OFF_W1);
  float*  H    = (float*)(w + OFF_H);
  __bf16* Xb1  = (__bf16*)(w + OFF_XB1);
  float*  sum0 = (float*)(w + OFF_ST);
  float*  sq0    = sum0 + 768;
  float*  scale0 = sq0 + 768;
  float*  shift0 = scale0 + 768;
  float*  sum1   = shift0 + 768;
  float*  sq1    = sum1 + 768;
  float*  scale1 = sq1 + 768;
  float*  shift1 = scale1 + 768;
  float*  meansS = (float*)(w + OFF_MEANS);
  int*    idx    = (int*)(w + OFF_IDX);

  // output sections
  float* outM   = (float*)d_out;                 // [B][E][OUT_T]
  float* outS   = outM + (size_t)B_ * E_ * OUT_T_;
  float* outDur = outS + (size_t)B_ * E_ * OUT_T_;
  float* outF0  = outDur + (size_t)B_ * T_;
  float* outEn  = outF0 + (size_t)B_ * T_;

  hipMemsetAsync(w + OFF_ST, 0, 24576, stream);

  // weight prepack
  pack_w<E_><<<(3 * C_ * E_ * 3 + 255) / 256, 256, 0, stream>>>(P_W0, Wp0);
  pack_w<C_><<<(3 * C_ * C_ * 3 + 255) / 256, 256, 0, stream>>>(P_W1, Wp1);

  // encodings -> padded t-major bf16
  zero_pad_rows<E_><<<(B_ * 2 * E_ + 255) / 256, 256, 0, stream>>>(Xb0, B_);
  transpose_bn_bf16<E_><<<dim3(T_ / 32, E_ / 32, B_), dim3(32, 8), 0, stream>>>(
      enc, Xb0, nullptr, nullptr, B_);

  // conv0 (E->C, K=3) + mish + BN stats
  conv_gemm_wmma<E_, 0><<<dim3(T_ / 64, C_ / 128, 3 * B_), 256, 0, stream>>>(
      Xb0, Wp0, P_b0, H, sum0, sq0);
  bn_finalize<<<3, 256, 0, stream>>>(sum0, sq0, P_g0, P_be0, scale0, shift0);

  // BN-fold + repack h0 -> padded t-major bf16
  zero_pad_rows<C_><<<(3 * B_ * 2 * C_ + 255) / 256, 256, 0, stream>>>(Xb1, 3 * B_);
  transpose_bn_bf16<C_><<<dim3(T_ / 32, C_ / 32, 3 * B_), dim3(32, 8), 0, stream>>>(
      H, Xb1, scale0, shift0, B_);

  // conv1 (C->C, K=3) + mish + BN stats (overwrites H)
  conv_gemm_wmma<C_, 1><<<dim3(T_ / 64, C_ / 128, 3 * B_), 256, 0, stream>>>(
      Xb1, Wp1, P_b1, H, sum1, sq1);
  bn_finalize<<<3, 256, 0, stream>>>(sum1, sq1, P_g1, P_be1, scale1, shift1);

  // pointwise conv -> the three prediction outputs
  final_pw<<<dim3(T_ / 256, B_, 3), 256, 0, stream>>>(
      H, scale1, shift1, P_Wp, P_bp, outDur, outF0, outEn);

  // means staging: enc + masked embedding convs
  emb_means<<<dim3(T_ / 256, E_, B_), 256, 0, stream>>>(
      enc, mask, logf0s, energies, f0W, f0b, enW, enb, meansS);

  // length regulation
  lr_idx<<<B_, 1024, 0, stream>>>(dur, idx);
  {
    size_t tot = (size_t)B_ * E_ * OUT_T_;
    lr_gather<<<(unsigned)((tot + 255) / 256), 256, 0, stream>>>(
        meansS, logstdp, idx, outM, outS);
  }
}
